// MultiHeadAttention_19310172963229
// MI455X (gfx1250) — matrix operand order repaired and checked by reading, not compiled, not run on hardware
//
#include <hip/hip_runtime.h>

// ---------------------------------------------------------------------------
// MI455X (gfx1250) fused multi-head attention, bf16 WMMA path.
//   - v_wmma_f32_16x16x32_bf16 everywhere (wave32, 16x16 tiles, K=32)
//   - register double-buffered GEMM (hide L2 latency behind WMMA)
//   - flash attention with a fully async, double-buffered tile pipeline:
//       K tile  : TDM tensor_load_to_lds (D# with LDS row padding)
//       V tile  : global_load_async_to_lds (raw), LDS->LDS transpose
//   - DPP16 butterfly reductions + exp2-domain online softmax
// ---------------------------------------------------------------------------

typedef __attribute__((ext_vector_type(16))) __bf16 v16bf;
typedef __attribute__((ext_vector_type(8)))  __bf16 v8bf;
typedef __attribute__((ext_vector_type(4)))  __bf16 v4bf;
typedef __attribute__((ext_vector_type(2)))  __bf16 v2bf;
typedef __attribute__((ext_vector_type(8)))  float  v8f;
typedef __attribute__((ext_vector_type(4)))  unsigned int u32x4;
typedef __attribute__((ext_vector_type(8)))  int i32x8;
typedef __attribute__((ext_vector_type(4)))  int i32x4;

static __device__ __forceinline__ v16bf bf_concat(v8bf lo, v8bf hi) {
    return __builtin_shufflevector(lo, hi, 0,1,2,3,4,5,6,7,8,9,10,11,12,13,14,15);
}

static __device__ __forceinline__ v8f wmma_bf16(v16bf a, v16bf b, v8f c) {
    return __builtin_amdgcn_wmma_f32_16x16x32_bf16(false, a, false, b, (short)0, c, false, false);
}

// ---- DPP16 16-lane butterfly reductions (VALU only) -----------------------
template <int CTRL>
static __device__ __forceinline__ float dppmov(float x) {
    union { float f; int i; } u, r;
    u.f = x;
    r.i = __builtin_amdgcn_mov_dpp(u.i, CTRL, 0xf, 0xf, true);
    return r.f;
}
static __device__ __forceinline__ float red16_max(float v) {
    v = fmaxf(v, dppmov<0xB1>(v));   // quad_perm [1,0,3,2]
    v = fmaxf(v, dppmov<0x4E>(v));   // quad_perm [2,3,0,1]
    v = fmaxf(v, dppmov<0x141>(v));  // row_half_mirror
    v = fmaxf(v, dppmov<0x140>(v));  // row_mirror
    return v;
}
static __device__ __forceinline__ float red16_sum(float v) {
    v += dppmov<0xB1>(v);
    v += dppmov<0x4E>(v);
    v += dppmov<0x141>(v);
    v += dppmov<0x140>(v);
    return v;
}

// ---- TDM: 2-D tile (32 rows x 64 bf16) global -> LDS, 72-elem LDS rows ----
// D# per CDNA5 ISA. pad: every 32 DWORDs (one
// 128B row) insert 4 DWORDs (16B) => LDS row stride 144B = 72 bf16 = KSTRIDE.
static __device__ __forceinline__ void tdm_load_k32x64(unsigned lds_addr,
                                                       const __bf16* gptr,
                                                       int rows_stride_elems) {
    unsigned long long ga = (unsigned long long)(uintptr_t)gptr;
    u32x4 g0;
    g0[0] = 1u;                                   // count=1 (valid), no gather
    g0[1] = lds_addr;                             // LDS byte address
    g0[2] = (unsigned)(ga & 0xFFFFFFFFu);         // global_addr[31:0]
    g0[3] = (unsigned)((ga >> 32) & 0x1FFFFFFu)   // global_addr[56:32]
          | (2u << 30);                           // type = 2 ("image")
    i32x8 g1;
    g1[0] = (1 << 16)      // data_size = 1 -> 2 bytes
          | (1 << 20)      // pad_enable
          | (4 << 22)      // pad_interval code 4 = 32 DWORDs
          | (3 << 25);     // pad_amount  code 3 = 4 DWORDs
    g1[1] = (64 & 0xFFFF) << 16;                  // tensor_dim0 = 64 (lo16)
    g1[2] = (2048 & 0xFFFF) << 16;                // tensor_dim1 = 2048 (lo16)
    g1[3] = (64 << 16);                           // tile_dim0 = 64
    g1[4] = 32;                                   // tile_dim1 = 32
    g1[5] = rows_stride_elems;                    // tensor_dim0_stride lo32
    g1[6] = 0;
    g1[7] = 0;
    i32x4 z4 = {0, 0, 0, 0};
    i32x8 z8 = {0, 0, 0, 0, 0, 0, 0, 0};
    // 6-arg toolchain variant: (g0, g1, g2, g3, g4, cpol)
    __builtin_amdgcn_tensor_load_to_lds(g0, g1, z4, z4, z8, 0);
}

// ---------------------------------------------------------------------------
// Kernel 0: f32 -> bf16 conversion (vectorized x4). n must be multiple of 4.
// ---------------------------------------------------------------------------
__global__ __launch_bounds__(256) void cvt_f32_bf16(const float* __restrict__ in,
                                                    __bf16* __restrict__ out, int n) {
    int stride = gridDim.x * blockDim.x;
    for (int j = blockIdx.x * blockDim.x + threadIdx.x; j < (n >> 2); j += stride) {
        float4 f = ((const float4*)in)[j];
        v4bf b;
        b.x = (__bf16)f.x; b.y = (__bf16)f.y; b.z = (__bf16)f.z; b.w = (__bf16)f.w;
        *(v4bf*)(out + 4 * (size_t)j) = b;
    }
}

// ---------------------------------------------------------------------------
// Kernel 1/3: C[row,col] = (sum_k A[row,k]*W[col,k] + bias[col]) * oscale
// One wave computes 64x64 (4x4 WMMA accs), K unrolled by 64, register
// ping-pong so next tile's global loads overlap current WMMAs.
// ---------------------------------------------------------------------------
template <typename OutT>
__global__ __launch_bounds__(32) void gemm_nt_bias(const __bf16* __restrict__ A,
                                                   const __bf16* __restrict__ W,
                                                   const float*  __restrict__ bias,
                                                   OutT* __restrict__ C,
                                                   int M, int Nn, int K, float oscale) {
    const int lane = threadIdx.x & 31;
    const int half = lane >> 4;
    const int lm   = lane & 15;
    const int tm = blockIdx.x * 64;
    const int tn = blockIdx.y * 64;

    const __bf16* Abase = A + (size_t)(tm + lm) * K;
    const __bf16* Wbase = W + (size_t)(tn + lm) * K;

    auto loadA = [&](v16bf (&a)[4], int k0) {
#pragma unroll
        for (int i = 0; i < 4; ++i) {
            const __bf16* ap = Abase + (size_t)i * 16 * K + k0;
            a[i] = bf_concat(*(const v8bf*)(ap + half * 8),
                             *(const v8bf*)(ap + 16 + half * 8));
        }
    };
    auto loadB = [&](v16bf (&b)[4], int k0) {
#pragma unroll
        for (int j = 0; j < 4; ++j) {
            const __bf16* wp = Wbase + (size_t)j * 16 * K + k0 + half * 8;
            b[j] = bf_concat(*(const v8bf*)(wp), *(const v8bf*)(wp + 16));
        }
    };

    v8f acc[4][4];
#pragma unroll
    for (int i = 0; i < 4; ++i)
#pragma unroll
        for (int j = 0; j < 4; ++j) acc[i][j] = (v8f){0,0,0,0,0,0,0,0};

    auto mma = [&](v16bf (&a)[4], v16bf (&b)[4]) {
#pragma unroll
        for (int i = 0; i < 4; ++i)
#pragma unroll
            for (int j = 0; j < 4; ++j)
                acc[i][j] = wmma_bf16(a[i], b[j], acc[i][j]);
    };

    v16bf a0[4], b0[4], a1[4], b1[4];
    loadA(a0, 0); loadB(b0, 0);
    for (int k0 = 0; k0 < K; k0 += 64) {            // K % 64 == 0
        loadA(a1, k0 + 32); loadB(b1, k0 + 32);
        mma(a0, b0);
        if (k0 + 64 < K) { loadA(a0, k0 + 64); loadB(b0, k0 + 64); }
        mma(a1, b1);
    }

#pragma unroll
    for (int j = 0; j < 4; ++j) {
        const int col = tn + j * 16 + lm;
        const float bj = bias[col];
#pragma unroll
        for (int i = 0; i < 4; ++i) {
#pragma unroll
            for (int r = 0; r < 8; ++r) {
                const int row = tm + i * 16 + r + 8 * half;
                C[(size_t)row * Nn + col] = (OutT)((acc[i][j][r] + bj) * oscale);
            }
        }
    }
}

// ---------------------------------------------------------------------------
// Kernel 2: flash attention. 256 threads = 8 waves x 16 q-rows; 32-key tiles.
// Double-buffered async pipeline:
//   stage(t+1): wave0 TDM-copies K tile, all waves async-copy raw V tile
//   while compute(t) runs on the other buffers.
// Q was pre-scaled by 0.125*log2(e) in its projection => softmax in exp2.
// ---------------------------------------------------------------------------
#define HD 64
#define KSTRIDE 72              // bf16 elems (144B = 16B-aligned, bank-skewed)
#define VRSTRIDE 72
#define PSTRIDE 40              // 80B rows
#define KTILE (32 * KSTRIDE)
#define VRTILE (32 * VRSTRIDE)

__global__ __launch_bounds__(256) void flash_attn(const __bf16* __restrict__ Q,
                                                  const __bf16* __restrict__ Km,
                                                  const __bf16* __restrict__ V,
                                                  __bf16* __restrict__ O,
                                                  int N, int D, int H) {
    const int bz   = blockIdx.z;
    const int h    = blockIdx.y;
    const int wave = threadIdx.x >> 5;
    const int lane = threadIdx.x & 31;
    const int half = lane >> 4;
    const int lm   = lane & 15;

    const size_t headoff = (size_t)bz * N * D + (size_t)h * HD;
    const __bf16* Qh = Q  + headoff;
    const __bf16* Kh = Km + headoff;
    const __bf16* Vh = V  + headoff;
    __bf16*       Oh = O  + headoff;

    const int qrow0 = blockIdx.x * 128 + wave * 16;

    __shared__ __bf16 kt_s[2 * KTILE];           // K tiles raw [key][d], padded rows
    __shared__ __bf16 vraw[2 * VRTILE];          // V tiles raw [key][d], padded rows
    __shared__ __bf16 vt[HD * PSTRIDE];          // V tile transposed [d][key]
    __shared__ __bf16 pl[8 * 16 * PSTRIDE];      // per-wave P staging [row][k]
    __bf16* pw = pl + wave * 16 * PSTRIDE;

    // async-V staging mapping: one b128 per thread (32 rows x 64 d)
    const int srow = threadIdx.x >> 3;           // 0..31 key
    const int sd0  = (threadIdx.x & 7) * 8;      // 0..56 d
    // LDS->LDS transpose mapping: two adjacent keys x 4 d per thread
    const int tk2 = (threadIdx.x & 15) * 2;      // 0..30 key pair
    const int td0 = (threadIdx.x >> 4) * 4;      // 0..60 d

    auto stage = [&](int kt, int buf) {
        unsigned vdst = (unsigned)(uintptr_t)&vraw[buf * VRTILE + srow * VRSTRIDE + sd0];
        unsigned long long vga =
            (unsigned long long)(uintptr_t)(Vh + (size_t)(kt + srow) * D + sd0);
        asm volatile("global_load_async_to_lds_b128 %0, %1, off"
                     :: "v"(vdst), "v"(vga) : "memory");
        if (wave == 0) {
            unsigned kdst = (unsigned)(uintptr_t)&kt_s[buf * KTILE];
            tdm_load_k32x64(kdst, Kh + (size_t)kt * D, D);
        }
    };

    // Q A-frags (Q already carries 0.125*log2e)
    v16bf aq[2];
#pragma unroll
    for (int ks = 0; ks < 2; ++ks) {
        const __bf16* qp = Qh + (size_t)(qrow0 + lm) * D + ks * 32;
        aq[ks] = bf_concat(*(const v8bf*)(qp + half * 8),
                           *(const v8bf*)(qp + 16 + half * 8));
    }

    v8f o[4];
#pragma unroll
    for (int dn = 0; dn < 4; ++dn) o[dn] = (v8f){0,0,0,0,0,0,0,0};
    float mrow[8], lrow[8];
#pragma unroll
    for (int r = 0; r < 8; ++r) { mrow[r] = -INFINITY; lrow[r] = 0.f; }

    // ---- prologue: stage tile 0 ----
    stage(0, 0);
    if (wave == 0) __builtin_amdgcn_s_wait_tensorcnt(0);
    asm volatile("s_wait_asynccnt 0x0" ::: "memory");
    __syncthreads();

    const int T = N / 32;
    for (int it = 0; it < T; ++it) {
        const int cur = it & 1;
        const int kt = it * 32;
        if (it + 1 < T) stage(kt + 32, cur ^ 1);     // overlap with compute(t)

        // ---- LDS->LDS transpose of current raw V tile -> vt ---------------
        {
            const __bf16* vr = vraw + cur * VRTILE;
            v4bf r0 = *(const v4bf*)(vr + tk2 * VRSTRIDE + td0);
            v4bf r1 = *(const v4bf*)(vr + (tk2 + 1) * VRSTRIDE + td0);
#pragma unroll
            for (int j = 0; j < 4; ++j) {
                v2bf p; p.x = r0[j]; p.y = r1[j];
                *(v2bf*)(vt + (td0 + j) * PSTRIDE + tk2) = p;
            }
        }
        __syncthreads();   // vt visible to all waves

        // ---- scores: S[16q x 32k] = Q(16x64).K^T, two 16-col WMMA tiles ---
        const __bf16* kcur = kt_s + cur * KTILE;
        v8f s[2];
#pragma unroll
        for (int c = 0; c < 2; ++c) {
            const __bf16* kp = kcur + (c * 16 + lm) * KSTRIDE;
            v8f acc = (v8f){0,0,0,0,0,0,0,0};
#pragma unroll
            for (int ks = 0; ks < 2; ++ks) {
                const __bf16* kpp = kp + ks * 32 + half * 8;
                acc = wmma_bf16(aq[ks],
                                bf_concat(*(const v8bf*)(kpp), *(const v8bf*)(kpp + 16)),
                                acc);
            }
            s[c] = acc;
        }

        // ---- online softmax in exp2 domain, DPP16 butterflies -------------
#pragma unroll
        for (int r = 0; r < 8; ++r) {
            const float s0 = s[0][r];
            const float s1 = s[1][r];
            const float tmax = red16_max(fmaxf(s0, s1));
            const float mnew = fmaxf(mrow[r], tmax);
            const float alpha = exp2f(mrow[r] - mnew);   // 0 on first tile
            const float p0 = exp2f(s0 - mnew);
            const float p1 = exp2f(s1 - mnew);
            s[0][r] = p0; s[1][r] = p1;
            lrow[r] = lrow[r] * alpha + red16_sum(p0 + p1);
            mrow[r] = mnew;
#pragma unroll
            for (int dn = 0; dn < 4; ++dn) o[dn][r] *= alpha;
        }

        // ---- restage P (C-layout) -> A-layout via per-wave LDS tile -------
#pragma unroll
        for (int c = 0; c < 2; ++c)
#pragma unroll
            for (int r = 0; r < 8; ++r)
                pw[(r + 8 * half) * PSTRIDE + c * 16 + lm] = (__bf16)s[c][r];
        asm volatile("s_wait_dscnt 0x0" ::: "memory");   // wave-local RAW fence

        v16bf ap;
        {
            const __bf16* pp = pw + lm * PSTRIDE;
            ap = bf_concat(*(const v8bf*)(pp + half * 8),
                           *(const v8bf*)(pp + 16 + half * 8));
        }

        // ---- O += P(16x32) . V(32x64) -------------------------------------
#pragma unroll
        for (int dn = 0; dn < 4; ++dn) {
            const __bf16* vp = vt + (dn * 16 + lm) * PSTRIDE + half * 8;
            o[dn] = wmma_bf16(ap,
                              bf_concat(*(const v8bf*)(vp), *(const v8bf*)(vp + 16)),
                              o[dn]);
        }

        // ---- end of iteration: staged tile t+1 must have landed -----------
        if (wave == 0) __builtin_amdgcn_s_wait_tensorcnt(0);
        asm volatile("s_wait_asynccnt 0x0" ::: "memory");
        __syncthreads();
    }

    // ---- epilogue: divide by l, store bf16 into [B,N,D] (merging heads) ----
    float inv[8];
#pragma unroll
    for (int r = 0; r < 8; ++r) inv[r] = 1.0f / lrow[r];
#pragma unroll
    for (int dn = 0; dn < 4; ++dn)
#pragma unroll
        for (int r = 0; r < 8; ++r) {
            const int q = qrow0 + r + 8 * half;
            const int d = dn * 16 + lm;
            Oh[(size_t)q * D + d] = (__bf16)(o[dn][r] * inv[r]);
        }
}

// ---------------------------------------------------------------------------
// Host launcher
// ---------------------------------------------------------------------------
extern "C" void kernel_launch(void* const* d_in, const int* in_sizes, int n_in,
                              void* d_out, int out_size, void* d_ws, size_t ws_size,
                              hipStream_t stream) {
    (void)in_sizes; (void)n_in; (void)out_size; (void)ws_size;
    constexpr int Bb = 2, N = 2048, D = 1024, H = 16;
    constexpr int M = Bb * N;
    const size_t MD = (size_t)M * D;
    const size_t DD = (size_t)D * D;

    const float* x  = (const float*)d_in[0];
    const float* Wq = (const float*)d_in[1];
    const float* bq = (const float*)d_in[2];
    const float* Wk = (const float*)d_in[3];
    const float* bk = (const float*)d_in[4];
    const float* Wv = (const float*)d_in[5];
    const float* bv = (const float*)d_in[6];
    const float* Wo = (const float*)d_in[7];
    const float* bo = (const float*)d_in[8];
    float* out = (float*)d_out;

    __bf16* w = (__bf16*)d_ws;
    __bf16* xb  = w; w += MD;
    __bf16* wqb = w; w += DD;
    __bf16* wkb = w; w += DD;
    __bf16* wvb = w; w += DD;
    __bf16* wob = w; w += DD;
    __bf16* qb  = w; w += MD;
    __bf16* kb  = w; w += MD;
    __bf16* vb  = w; w += MD;
    __bf16* ab  = w; w += MD;

    cvt_f32_bf16<<<1024, 256, 0, stream>>>(x,  xb,  (int)MD);
    cvt_f32_bf16<<<256,  256, 0, stream>>>(Wq, wqb, (int)DD);
    cvt_f32_bf16<<<256,  256, 0, stream>>>(Wk, wkb, (int)DD);
    cvt_f32_bf16<<<256,  256, 0, stream>>>(Wv, wvb, (int)DD);
    cvt_f32_bf16<<<256,  256, 0, stream>>>(Wo, wob, (int)DD);

    // Q carries softmax scale in log2 domain: 1/sqrt(64) * log2(e)
    const float qscale = 0.125f * 1.4426950408889634f;
    dim3 ggrid(M / 64, D / 64);
    gemm_nt_bias<__bf16><<<ggrid, 32, 0, stream>>>(xb, wqb, bq, qb, M, D, D, qscale);
    gemm_nt_bias<__bf16><<<ggrid, 32, 0, stream>>>(xb, wkb, bk, kb, M, D, D, 1.0f);
    gemm_nt_bias<__bf16><<<ggrid, 32, 0, stream>>>(xb, wvb, bv, vb, M, D, D, 1.0f);

    dim3 agrid(N / 128, H, Bb);
    flash_attn<<<agrid, 256, 0, stream>>>(qb, kb, vb, ab, N, D, H);

    gemm_nt_bias<float><<<ggrid, 32, 0, stream>>>(ab, wob, bo, out, M, D, D, 1.0f);
}
